// MoE_80333068304662
// MI455X (gfx1250) — matrix operand order repaired and checked by reading, not compiled, not run on hardware
//
#include <hip/hip_runtime.h>
#include <hip/hip_bf16.h>

// ---------------- problem constants (match reference) ----------------
#define T_TOK 2048   // B*S tokens
#define DIM   1024   // D
#define NEXP  16     // E
#define HID   2048   // H
#define MTILE 32     // tokens per workgroup tile (2 WMMA M-tiles)
#define TILES_PER_E 64   // T_TOK / MTILE worst-case tiles per expert
#define HCHUNK 256   // hidden-dim chunk fused through LDS

// ---------------- CDNA5 WMMA types ----------------
typedef __attribute__((ext_vector_type(16))) __bf16   v16bf;
typedef __attribute__((ext_vector_type(2)))  __bf16   v2bf;
typedef __attribute__((ext_vector_type(8)))  float    v8f;
typedef __attribute__((ext_vector_type(8)))  unsigned int v8u;

// Packed f32x2 -> bf16x2 (RNE) in ONE VALU op: v_cvt_pk_bf16_f32.
static __device__ __forceinline__ unsigned int cvt_pk_bf16(float lo, float hi) {
#if __has_builtin(__builtin_amdgcn_cvt_pk_bf16_f32)
    v2bf v = __builtin_amdgcn_cvt_pk_bf16_f32(lo, hi);
    return __builtin_bit_cast(unsigned int, v);
#else
    unsigned int r;
    asm("v_cvt_pk_bf16_f32 %0, %1, %2" : "=v"(r) : "v"(lo), "v"(hi));
    return r;
#endif
}
static __device__ __forceinline__ unsigned short cvt_bf16(float f) {
    return (unsigned short)(cvt_pk_bf16(f, f) & 0xffffu);
}
static __device__ __forceinline__ float gelu_exact(float v) {
    return 0.5f * v * (1.0f + erff(v * 0.70710678118654752440f));
}

static __device__ __forceinline__ v8f wmma_bf16(v8u a, v8u b, v8f c) {
    return __builtin_amdgcn_wmma_f32_16x16x32_bf16(
        false, __builtin_bit_cast(v16bf, a),
        false, __builtin_bit_cast(v16bf, b),
        (short)0, c, false, false);
}

// A-fragment (16x32 bf16, M=token) from LDS, row-major with padded stride.
// Lanes 0-15: M=lane, halves K0..7 (v0:3) and K16..23 (v4:7);
// lanes 16-31: M=lane-16, K8..15 and K24..31.  (ISA 7.12.2 16-bit A layout)
static __device__ __forceinline__ v8u load_a_frag(const unsigned short* base,
                                                  int stride_h, int row16,
                                                  int hsel, int kk) {
    const unsigned short* p = base + row16 * stride_h + kk + hsel * 8;
    uint4 lo = *(const uint4*)p;          // ds_load_b128
    uint4 hi = *(const uint4*)(p + 16);   // K + 16
    v8u f;
    f[0] = lo.x; f[1] = lo.y; f[2] = lo.z; f[3] = lo.w;
    f[4] = hi.x; f[5] = hi.y; f[6] = hi.z; f[7] = hi.w;
    return f;
}

// B-fragment (32x16): lane = N column (lane%16); elements K0..7,K16..23 (lanes 0-15)
// or K8..15,K24..31 (lanes 16-31), strided by ld in the row-major weight matrix;
// conversion is 8x v_cvt_pk_bf16_f32.
static __device__ __forceinline__ v8u load_b_frag_f32(const float* p, int ld) {
    v8u r;
#pragma unroll
    for (int j = 0; j < 4; ++j) {
        r[j]     = cvt_pk_bf16(p[(2 * j) * ld], p[(2 * j + 1) * ld]);
        r[4 + j] = cvt_pk_bf16(p[(16 + 2 * j) * ld], p[(17 + 2 * j) * ld]);
    }
    return r;
}

// ---------------- kernel 1: zero expert counters ----------------
__global__ void moe_zero_counts(int* counts) {
    if (threadIdx.x < NEXP) counts[threadIdx.x] = 0;
}

// ---------------- kernel 2: router (top-2, analytic combine wts) ----------------
__global__ void moe_router(const float* __restrict__ x,
                           const float* __restrict__ Wr,
                           const float* __restrict__ br,
                           int* __restrict__ counts,
                           int* __restrict__ toks,
                           float* __restrict__ wts) {
    int t = blockIdx.x * blockDim.x + threadIdx.x;
    if (t >= T_TOK) return;
    float acc[NEXP];
#pragma unroll
    for (int e = 0; e < NEXP; ++e) acc[e] = br[e];
    const float* xr = x + (size_t)t * DIM;
    for (int d = 0; d < DIM; ++d) {
        float xv = xr[d];
        const float* wrow = Wr + (size_t)d * NEXP;
#pragma unroll
        for (int e = 0; e < NEXP; ++e) acc[e] = fmaf(xv, wrow[e], acc[e]);
    }
    // top-2 (first-max wins ties, matching top_k's lowest-index preference)
    int i0 = 0; float l0 = acc[0];
#pragma unroll
    for (int e = 1; e < NEXP; ++e) { if (acc[e] > l0) { l0 = acc[e]; i0 = e; } }
    int i1 = -1; float l1 = -3.4e38f;
#pragma unroll
    for (int e = 0; e < NEXP; ++e) { if (e != i0 && acc[e] > l1) { l1 = acc[e]; i1 = e; } }
    // softmax-renormalized top-2 weights: denominators cancel analytically
    float w0 = 1.0f / (1.0f + expf(l1 - l0));
    float w1 = 1.0f - w0;
    int p0 = atomicAdd(&counts[i0], 1);
    toks[i0 * T_TOK + p0] = t; wts[i0 * T_TOK + p0] = w0;
    int p1 = atomicAdd(&counts[i1], 1);
    toks[i1 * T_TOK + p1] = t; wts[i1 * T_TOK + p1] = w1;
}

// ---------------- kernel 3: residual init out = x ----------------
__global__ void moe_residual_init(const float* __restrict__ x,
                                  float* __restrict__ out, int n) {
    for (int i = blockIdx.x * blockDim.x + threadIdx.x; i < n;
         i += gridDim.x * blockDim.x)
        out[i] = x[i];
}

// ---------------- kernel 4: fused expert MLP (bf16 WMMA, f32 accum) ----------------
// 32-token M-tile: each B-fragment (16 loads + 8 cvt) feeds 2 WMMAs, halving
// weight bandwidth and conversion ALU per FLOP vs a 16-token tile.
__global__ __launch_bounds__(256)
void moe_expert_gemm(const float* __restrict__ x,
                     const float* __restrict__ W1, const float* __restrict__ b1,
                     const float* __restrict__ W2, const float* __restrict__ b2,
                     const int* __restrict__ counts,
                     const int* __restrict__ toks,
                     const float* __restrict__ wts,
                     float* __restrict__ out) {
    const int e    = blockIdx.x >> 6;        // / TILES_PER_E
    const int tile = blockIdx.x & (TILES_PER_E - 1);
    const int n_tok = counts[e];
    if (tile * MTILE >= n_tok) return;       // uniform early-exit

    // padded LDS rows (stride 1032/264 halves = 16B-aligned, bank-spread)
    // 32x1032x2 + 32x264x2 + 256 = ~83 KB of the WGP's 320 KB LDS.
    __shared__ __align__(16) unsigned short Xs[MTILE][1032];  // 32x1024 bf16 X
    __shared__ __align__(16) unsigned short Hs[MTILE][264];   // 32x256  bf16 h
    __shared__ int   tok_s[MTILE];
    __shared__ float wt_s[MTILE];

    const int tid   = threadIdx.x;
    const int lane  = tid & 31;
    const int wave  = tid >> 5;
    const int row16 = lane & 15;
    const int hsel  = lane >> 4;

    if (tid < MTILE) {
        int idx = tile * MTILE + tid;
        if (idx < n_tok) {
            tok_s[tid] = toks[e * T_TOK + idx];
            wt_s[tid]  = wts[e * T_TOK + idx];
        } else {                              // pad row: weight 0 kills output
            tok_s[tid] = toks[e * T_TOK];
            wt_s[tid]  = 0.0f;
        }
    }
    __syncthreads();

    // stage X tile -> LDS as bf16: float2 load -> v_cvt_pk_bf16_f32 -> b32 store
    for (int i = tid; i < MTILE * (DIM / 2); i += 256) {
        int r = i >> 9, c2 = (i & (DIM / 2 - 1)) * 2;
        const float2 v = *(const float2*)(x + (size_t)tok_s[r] * DIM + c2);
        *(unsigned int*)&Xs[r][c2] = cvt_pk_bf16(v.x, v.y);
    }
    __syncthreads();

    const float* W1e = W1 + (size_t)e * DIM * HID;
    const float* W2e = W2 + (size_t)e * HID * DIM;
    const float* b1e = b1 + (size_t)e * HID;
    const float* b2e = b2 + (size_t)e * DIM;

    const int yn0 = wave * 128;              // this wave's 128 output columns
    v8f yacc0[8], yacc1[8];                  // M rows 0-15 / 16-31
#pragma unroll
    for (int t = 0; t < 8; ++t) {
        yacc0[t] = (v8f){0.f, 0.f, 0.f, 0.f, 0.f, 0.f, 0.f, 0.f};
        yacc1[t] = (v8f){0.f, 0.f, 0.f, 0.f, 0.f, 0.f, 0.f, 0.f};
    }

    for (int hc = 0; hc < HID; hc += HCHUNK) {
        // ---- fc1: wave computes h columns [hc + wave*32, +32), both M halves
        const int n0 = hc + wave * 32;
        v8f a00 = (v8f){0.f,0.f,0.f,0.f,0.f,0.f,0.f,0.f};
        v8f a01 = a00, a10 = a00, a11 = a00;
#pragma unroll 2
        for (int kk = 0; kk < DIM; kk += 32) {
            if (kk + 32 < DIM)
                __builtin_prefetch(W1e + (size_t)(kk + 32 + lane) * HID + n0, 0, 1);
            v8u af0 = load_a_frag(&Xs[0][0],       1032, row16, hsel, kk);
            v8u af1 = load_a_frag(&Xs[16][0],      1032, row16, hsel, kk);
            v8u bf0 = load_b_frag_f32(W1e + (size_t)(kk + hsel * 8) * HID + n0 + row16, HID);
            v8u bf1 = load_b_frag_f32(W1e + (size_t)(kk + hsel * 8) * HID + n0 + 16 + row16, HID);
            a00 = wmma_bf16(af0, bf0, a00);
            a10 = wmma_bf16(af1, bf0, a10);
            a01 = wmma_bf16(af0, bf1, a01);
            a11 = wmma_bf16(af1, bf1, a11);
        }
        // bias + exact GELU -> bf16 h chunk in LDS
        float bias0 = b1e[n0 + row16];
        float bias1 = b1e[n0 + 16 + row16];
#pragma unroll
        for (int r = 0; r < 8; ++r) {
            int m0 = hsel * 8 + r;           // C layout: lanes16-31 hold M=8..15
            int m1 = m0 + 16;
            Hs[m0][(n0 - hc) + row16]      = cvt_bf16(gelu_exact(a00[r] + bias0));
            Hs[m0][(n0 - hc) + 16 + row16] = cvt_bf16(gelu_exact(a01[r] + bias1));
            Hs[m1][(n0 - hc) + row16]      = cvt_bf16(gelu_exact(a10[r] + bias0));
            Hs[m1][(n0 - hc) + 16 + row16] = cvt_bf16(gelu_exact(a11[r] + bias1));
        }
        __syncthreads();

        // ---- fc2 partial: Y[:, yn0:yn0+128] += gelu_chunk @ W2[hc:hc+256, :]
#pragma unroll 1
        for (int kk = 0; kk < HCHUNK; kk += 32) {
            v8u af0 = load_a_frag(&Hs[0][0],  264, row16, hsel, kk);
            v8u af1 = load_a_frag(&Hs[16][0], 264, row16, hsel, kk);
            const float* w2row = W2e + (size_t)(hc + kk + hsel * 8) * DIM + yn0 + row16;
#pragma unroll
            for (int t = 0; t < 8; ++t) {
                v8u bf = load_b_frag_f32(w2row + t * 16, DIM);
                yacc0[t] = wmma_bf16(af0, bf, yacc0[t]);
                yacc1[t] = wmma_bf16(af1, bf, yacc1[t]);
            }
        }
        __syncthreads();                     // Hs reused next chunk
    }

    // epilogue: bias, combine weight, atomic accumulate into residual output
#pragma unroll
    for (int t = 0; t < 8; ++t) {
        int n = yn0 + t * 16 + row16;
        float bias = b2e[n];
#pragma unroll
        for (int r = 0; r < 8; ++r) {
            int m0 = hsel * 8 + r;
            int m1 = m0 + 16;
            float v0 = (yacc0[t][r] + bias) * wt_s[m0];
            float v1 = (yacc1[t][r] + bias) * wt_s[m1];
            atomicAdd(&out[(size_t)tok_s[m0] * DIM + n], v0);
            atomicAdd(&out[(size_t)tok_s[m1] * DIM + n], v1);
        }
    }
}

// ---------------- host-side launch ----------------
extern "C" void kernel_launch(void* const* d_in, const int* in_sizes, int n_in,
                              void* d_out, int out_size, void* d_ws, size_t ws_size,
                              hipStream_t stream) {
    const float* x  = (const float*)d_in[0];
    const float* Wr = (const float*)d_in[1];
    const float* br = (const float*)d_in[2];
    const float* W1 = (const float*)d_in[3];
    const float* b1 = (const float*)d_in[4];
    const float* W2 = (const float*)d_in[5];
    const float* b2 = (const float*)d_in[6];
    float* out = (float*)d_out;

    char* ws = (char*)d_ws;
    int*   counts = (int*)ws;                                    // 16 ints
    int*   toks   = (int*)(ws + 256);                            // E*T ints
    float* wts    = (float*)(ws + 256 + sizeof(int) * NEXP * T_TOK);

    moe_zero_counts<<<1, 32, 0, stream>>>(counts);
    moe_router<<<T_TOK / 256, 256, 0, stream>>>(x, Wr, br, counts, toks, wts);
    moe_residual_init<<<512, 256, 0, stream>>>(x, out, T_TOK * DIM);
    moe_expert_gemm<<<NEXP * TILES_PER_E, 256, 0, stream>>>(
        x, W1, b1, W2, b2, counts, toks, wts, out);
}
